// HMAGT_88192858456455
// MI455X (gfx1250) — matrix operand order repaired and checked by reading, not compiled, not run on hardware
//
#include <hip/hip_runtime.h>
#include <math.h>

// ---------------------------------------------------------------------------
// CDNA5 (gfx1250) wave32 WMMA types
// ---------------------------------------------------------------------------
typedef __attribute__((ext_vector_type(16))) __bf16        v16bf;
typedef __attribute__((ext_vector_type(8)))  float         v8f;
typedef __attribute__((ext_vector_type(4)))  unsigned int  v4u;

union Frag16 {
  v16bf          v;
  unsigned short u[16];
  v4u            q[2];
};

__device__ __forceinline__ unsigned short f2bf(float f) {
  unsigned int u = __float_as_uint(f);
  unsigned int r = u + 0x7FFFu + ((u >> 16) & 1u);   // round-to-nearest-even
  return (unsigned short)(r >> 16);
}
__device__ __forceinline__ float bf2f(unsigned short h) {
  return __uint_as_float(((unsigned int)h) << 16);
}

// ---------------------------------------------------------------------------
// 0) f32 -> bf16 tensor conversion
// ---------------------------------------------------------------------------
__global__ void k_tobf16(const float* __restrict__ s, unsigned short* __restrict__ d,
                         int n) {
  int i = blockIdx.x * blockDim.x + threadIdx.x;
  if (i < n) d[i] = f2bf(s[i]);
}

// ---------------------------------------------------------------------------
// 1) conv1(3->64, 3x3, pad1) + BN + ReLU + maxpool2 fused.  K=27 too skinny
//    for WMMA; branch-free VALU (clamped loads + select), bf16 feat1 out.
// ---------------------------------------------------------------------------
__global__ void k_conv1_bn_relu_pool(const float* __restrict__ x,
                                     const float* __restrict__ w,
                                     const float* __restrict__ cb,
                                     const float* __restrict__ g,
                                     const float* __restrict__ bb,
                                     unsigned short* __restrict__ feat1) {
  int idx = blockIdx.x * blockDim.x + threadIdx.x;    // 16*64*128*128
  if (idx >= 16 * 64 * 128 * 128) return;
  int xo = idx & 127, yo = (idx >> 7) & 127, c = (idx >> 14) & 63, b = idx >> 20;
  const float* xb = x + (size_t)b * 3 * 256 * 256;
  const float* wc = w + c * 27;
  float bias = cb[c];
  float s    = g[c] * rsqrtf(1.0f + 1e-5f);
  float off  = bb[c];
  float best = -__builtin_inff();
  for (int dy = 0; dy < 2; ++dy)
    for (int dx = 0; dx < 2; ++dx) {
      int Y = 2 * yo + dy, X = 2 * xo + dx;
      float acc = bias;
      for (int ic = 0; ic < 3; ++ic)
#pragma unroll
        for (int ky = 0; ky < 3; ++ky)
#pragma unroll
          for (int kx = 0; kx < 3; ++kx) {
            int yy = Y + ky - 1, xx = X + kx - 1;
            bool ok = ((unsigned)yy < 256u) && ((unsigned)xx < 256u);
            int yc = ok ? yy : 0, xc = ok ? xx : 0;
            float xv = xb[ic * 65536 + (yc << 8) + xc];
            acc += wc[ic * 9 + ky * 3 + kx] * (ok ? xv : 0.0f);
          }
      best = fmaxf(best, fmaxf(acc * s + off, 0.0f));
    }
  feat1[idx] = f2bf(best);
}

// ---------------------------------------------------------------------------
// 2) conv2(64->64, 3x3, pad1) implicit GEMM via v_wmma_f32_16x16x32_bf16.
//    One wave = 16 pixels x 64 out-channels; K=576 in 18 chunks of 32.
//    Weight chunk (64oc x 32k = 4KB) staged cooperatively in LDS, shared by
//    the 4 waves of the block; A = branch-free im2col gather.
// ---------------------------------------------------------------------------
__global__ void k_conv2_wmma(const unsigned short* __restrict__ feat1,
                             const unsigned short* __restrict__ w2,
                             const float* __restrict__ cb,
                             const float* __restrict__ g,
                             const float* __restrict__ bb,
                             unsigned short* __restrict__ out) {
  __shared__ unsigned short sw[64 * 32];              // 4 KB weight chunk
  int gw   = (blockIdx.x * blockDim.x + threadIdx.x) >> 5;  // [0,16384)
  int lane = threadIdx.x & 31;
  int p0    = gw << 4;
  int b     = p0 >> 14;                               // 16384 pixels / image
  int pbase = p0 & 16383;
  const unsigned short* img = feat1 + (size_t)b * 64 * 16384;

  int m   = lane & 15;
  int kb  = (lane >> 4) << 3;                         // A: K split 0/8 (+16)
  int kb2 = (lane >> 4) << 4;                         // unused: B reads at kb, K split 0/8 (+16)
  int n   = lane & 15;
  int pix = pbase + m;
  int py  = pix >> 7, px = pix & 127;

  int tt   = threadIdx.x;
  int soc  = tt >> 1;                                 // staging: oc per 2 thr
  int shal = (tt & 1) << 4;                           // 0 or 16

  v8f acc[4] = {};
  for (int kk = 0; kk < 576; kk += 32) {
    // cooperative weight staging: 64 x 32 bf16 chunk
    {
      const unsigned short* src = w2 + (size_t)soc * 576 + kk + shal;
      v4u* dq = (v4u*)(sw + soc * 32 + shal);
      dq[0] = *(const v4u*)src;
      dq[1] = *(const v4u*)(src + 8);
      if (kk + 32 < 576) __builtin_prefetch(src + 32, 0, 0);
    }
    __syncthreads();
    Frag16 fa;
#pragma unroll
    for (int i = 0; i < 16; ++i) {
      int k  = kk + ((i < 8) ? (kb + i) : (8 + kb + i));   // ISA A 16x32 layout
      int ic = k / 9;
      int r  = k - ic * 9;
      int ky = r / 3, kx = r - (r / 3) * 3;
      int yy = py + ky - 1, xx = px + kx - 1;
      bool ok = ((unsigned)yy < 128u) && ((unsigned)xx < 128u);
      int yc = ok ? yy : 0, xc = ok ? xx : 0;
      unsigned short t = img[ic * 16384 + (yc << 7) + xc];
      fa.u[i] = ok ? t : (unsigned short)0;
    }
#pragma unroll
    for (int gq = 0; gq < 4; ++gq) {
      Frag16 fb;
      const unsigned short* sp = sw + (gq * 16 + n) * 32 + kb;
      fb.q[0] = *(const v4u*)sp;
      fb.q[1] = *(const v4u*)(sp + 16);
      acc[gq] = __builtin_amdgcn_wmma_f32_16x16x32_bf16(false, fa.v, false, fb.v,
                                                        (short)0, acc[gq], false, false);
    }
    __syncthreads();
  }
  int mbase = (lane >> 4) << 3;
  float rs  = rsqrtf(1.0f + 1e-5f);
#pragma unroll
  for (int gq = 0; gq < 4; ++gq) {
    int oc = gq * 16 + n;
    float sc = g[oc] * rs, of = bb[oc], cbias = cb[oc];
    size_t basep = ((size_t)b * 64 + oc) * 16384 + pbase + mbase;
#pragma unroll
    for (int i = 0; i < 8; ++i)
      out[basep + i] = f2bf(fmaxf((acc[gq][i] + cbias) * sc + of, 0.0f));
  }
}

// ---------------------------------------------------------------------------
// 3) maxpool2 on conv2 output -> feat2 bf16 [16][64][64][64]
// ---------------------------------------------------------------------------
__global__ void k_pool2(const unsigned short* __restrict__ in,
                        unsigned short* __restrict__ out) {
  int idx = blockIdx.x * blockDim.x + threadIdx.x;    // 16*64*64*64
  if (idx >= 16 * 64 * 64 * 64) return;
  int xo = idx & 63, yo = (idx >> 6) & 63, bc = idx >> 12;
  const unsigned short* p = in + (size_t)bc * 16384;
  float a = bf2f(p[(2 * yo) * 128 + 2 * xo]);
  a = fmaxf(a, bf2f(p[(2 * yo) * 128 + 2 * xo + 1]));
  a = fmaxf(a, bf2f(p[(2 * yo + 1) * 128 + 2 * xo]));
  a = fmaxf(a, bf2f(p[(2 * yo + 1) * 128 + 2 * xo + 1]));
  out[idx] = f2bf(a);
}

// ---------------------------------------------------------------------------
// 4) patch extraction + projection (3600x4096 @ 4096x128) via WMMA.
//    One wave = 16 patch rows x 128 cols.  Per-K-chunk weight staging in LDS
//    (128 cols x 32 k = 8KB) shared by all 4 waves; A gathered from feat2.
// ---------------------------------------------------------------------------
__global__ void k_proj_wmma(const unsigned short* __restrict__ feat2,
                            const unsigned short* __restrict__ wproj,
                            const float* __restrict__ bias,
                            float* __restrict__ nodes) {
  __shared__ unsigned short sw[128 * 32];             // 8 KB weight chunk
  int gw = (blockIdx.x * blockDim.x + threadIdx.x) >> 5;
  bool active = (gw < 225);                           // wave-uniform
  int lane = threadIdx.x & 31;
  int m = lane & 15, n = lane & 15;
  int kb = (lane >> 4) << 3, kb2 = (lane >> 4) << 4;
  int row  = active ? (gw * 16 + m) : 0;              // [0,3600)
  int b    = row / 225, node = row - b * 225;
  int ph   = node / 15,  pw  = node - ph * 15;
  const unsigned short* img = feat2 + (size_t)b * 64 * 4096;
  int ybase = ph * 4, xbase = pw * 4;
  int scol = threadIdx.x;                             // staging col per thread

  v8f acc[8] = {};
  for (int kk = 0; kk < 4096; kk += 32) {
    {
      const unsigned short* src = wproj + (size_t)scol * 4096 + kk;
      v4u* dq = (v4u*)(sw + scol * 32);
      dq[0] = *(const v4u*)(src + 0);
      dq[1] = *(const v4u*)(src + 8);
      dq[2] = *(const v4u*)(src + 16);
      dq[3] = *(const v4u*)(src + 24);
      if (kk + 32 < 4096) __builtin_prefetch(src + 32, 0, 0);
    }
    __syncthreads();
    if (active) {
      Frag16 fa;
#pragma unroll
      for (int i = 0; i < 16; ++i) {
        int k  = kk + ((i < 8) ? (kb + i) : (8 + kb + i));
        int c  = k >> 6, py = (k >> 3) & 7, px = k & 7;  // feat = c*64+py*8+px
        fa.u[i] = img[c * 4096 + ((ybase + py) << 6) + xbase + px];
      }
#pragma unroll
      for (int gq = 0; gq < 8; ++gq) {
        Frag16 fb;
        const unsigned short* sp = sw + (gq * 16 + n) * 32 + kb;
        fb.q[0] = *(const v4u*)sp;
        fb.q[1] = *(const v4u*)(sp + 16);
        acc[gq] = __builtin_amdgcn_wmma_f32_16x16x32_bf16(false, fa.v, false, fb.v,
                                                          (short)0, acc[gq], false, false);
      }
    }
    __syncthreads();
  }
  if (!active) return;
  int mbase = (lane >> 4) << 3;
#pragma unroll
  for (int gq = 0; gq < 8; ++gq) {
    int col = gq * 16 + n;
    float bv = bias[col];
#pragma unroll
    for (int i = 0; i < 8; ++i)
      nodes[(size_t)(gw * 16 + mbase + i) * 128 + col] = acc[gq][i] + bv;
  }
}

// ---------------------------------------------------------------------------
// 5) generic bf16 WMMA linear: dst[r][o] = act(src[r]·W[o] + b[o]) (+res)
//    One wave computes a 16x16 tile; rows%16==0, in_dim%32==0, out_dim%16==0.
//    Optional bf16 mirror output for chaining into the next WMMA layer.
// ---------------------------------------------------------------------------
__global__ void k_linear_wmma(const unsigned short* __restrict__ A,
                              const unsigned short* __restrict__ W,
                              const float* __restrict__ bias,
                              const float* __restrict__ residual,
                              float* __restrict__ dst,
                              unsigned short* __restrict__ dst_bf,
                              int rows, int in_dim, int out_dim, int relu) {
  int gw = (blockIdx.x * blockDim.x + threadIdx.x) >> 5;
  int colTiles = out_dim >> 4;
  int tiles = (rows >> 4) * colTiles;
  if (gw >= tiles) return;                            // wave-uniform
  int ct = gw % colTiles, rt = gw / colTiles;
  int lane = threadIdx.x & 31;
  int m = lane & 15, n = lane & 15;
  int kb = (lane >> 4) << 3, kb2 = (lane >> 4) << 4;
  int row = rt * 16 + m;

  v8f acc = {};
  for (int kk = 0; kk < in_dim; kk += 32) {
    Frag16 fa, fb;
    const unsigned short* ap = A + (size_t)row * in_dim + kk;
    fa.q[0] = *(const v4u*)(ap + kb);
    fa.q[1] = *(const v4u*)(ap + 16 + kb);
    const unsigned short* wp = W + (size_t)(ct * 16 + n) * in_dim + kk + kb;
    fb.q[0] = *(const v4u*)wp;
    fb.q[1] = *(const v4u*)(wp + 16);
    acc = __builtin_amdgcn_wmma_f32_16x16x32_bf16(false, fa.v, false, fb.v,
                                                  (short)0, acc, false, false);
  }
  int mbase = (lane >> 4) << 3;
  int col = ct * 16 + n;
  float bv = bias ? bias[col] : 0.0f;
#pragma unroll
  for (int i = 0; i < 8; ++i) {
    int r = rt * 16 + mbase + i;
    float v = acc[i] + bv;
    if (relu) v = fmaxf(v, 0.0f);
    if (residual) v += residual[(size_t)r * out_dim + col];
    dst[(size_t)r * out_dim + col] = v;
    if (dst_bf) dst_bf[(size_t)r * out_dim + col] = f2bf(v);
  }
}

// ---------------------------------------------------------------------------
// 6) adjacency (static 15x15 grid): 8-NN with stable (d^2, index) order,
//    then adj2 = (adj1 @ adj1 > 0) with zero diagonal.
// ---------------------------------------------------------------------------
__global__ void k_adj1(float* __restrict__ adj1) {
  int i = threadIdx.x;
  if (i >= 225) return;
  for (int j = 0; j < 225; ++j) adj1[i * 225 + j] = 0.0f;
  int ii = i / 15, jj = i - (i / 15) * 15;
  int prev = -1;                                      // lexicographic key
  for (int sel = 0; sel < 8; ++sel) {
    int bestKey = 1 << 30;
    for (int j = 0; j < 225; ++j) {
      if (j == i) continue;
      int dy = ii - j / 15, dx = jj - (j - (j / 15) * 15);
      int key = (dy * dy + dx * dx) * 256 + j;
      if (key > prev && key < bestKey) bestKey = key;
    }
    prev = bestKey;
    adj1[i * 225 + (bestKey & 255)] = 1.0f;
  }
}

__global__ void k_adj2(const float* __restrict__ adj1, float* __restrict__ adj2) {
  int i = blockIdx.x, j = threadIdx.x;
  if (j >= 225) return;
  float v = 0.0f;
  if (j != i) {
    for (int k = 0; k < 225; ++k)
      if (adj1[i * 225 + k] != 0.0f && adj1[k * 225 + j] != 0.0f) { v = 1.0f; break; }
  }
  adj2[i * 225 + j] = v;
}

// ---------------------------------------------------------------------------
// 7) graph aggregation: dst = adj @ nodes (adj entries 0/1), output bf16
// ---------------------------------------------------------------------------
__global__ void k_aggregate(const float* __restrict__ adj,
                            const float* __restrict__ nodes,
                            unsigned short* __restrict__ dst) {
  int idx = blockIdx.x * blockDim.x + threadIdx.x;    // 16*225*128
  if (idx >= 16 * 225 * 128) return;
  int d = idx & 127, r = idx >> 7;
  int b = r / 225, i = r - b * 225;
  const float* arow = adj + i * 225;
  const float* nb   = nodes + (size_t)b * 225 * 128 + d;
  float s = 0.0f;
  for (int j = 0; j < 225; ++j)
    if (arow[j] != 0.0f) s += nb[(size_t)j * 128];
  dst[idx] = f2bf(s);
}

// ---------------------------------------------------------------------------
// 8) gating: g = sigmoid([z1,z2] @ gate_w^T + gate_b); z = g*z1 + (1-g)*z2
// ---------------------------------------------------------------------------
__global__ void k_gate(const float* __restrict__ z1, const float* __restrict__ z2,
                       const float* __restrict__ gwt, const float* __restrict__ gb,
                       float* __restrict__ z) {
  int idx = blockIdx.x * blockDim.x + threadIdx.x;    // 16*225*128
  if (idx >= 16 * 225 * 128) return;
  int d = idx & 127, r = idx >> 7;
  const float* z1r = z1 + (size_t)r * 128;
  const float* z2r = z2 + (size_t)r * 128;
  const float* wr  = gwt + (size_t)d * 256;
  float t = gb[d];
  for (int e = 0; e < 128; ++e) t += z1r[e] * wr[e];
  for (int e = 0; e < 128; ++e) t += z2r[e] * wr[128 + e];
  float gt = 1.0f / (1.0f + expf(-t));
  z[idx] = gt * z1r[d] + (1.0f - gt) * z2r[d];
}

// ---------------------------------------------------------------------------
// 9) small VALU linear (scores out_dim=1, final classifier out_dim=2)
// ---------------------------------------------------------------------------
__global__ void k_linear(const float* __restrict__ src, const float* __restrict__ W,
                         const float* __restrict__ bias, float* __restrict__ dst,
                         int rows, int in_dim, int out_dim) {
  int idx = blockIdx.x * blockDim.x + threadIdx.x;
  if (idx >= rows * out_dim) return;
  int o = idx % out_dim, r = idx / out_dim;
  const float* s = src + (size_t)r * in_dim;
  const float* w = W + (size_t)o * in_dim;
  float acc = bias[o];
  for (int e = 0; e < in_dim; ++e) acc += s[e] * w[e];
  dst[idx] = acc;
}

// ---------------------------------------------------------------------------
// 10) stable top-k (k=112) pooling: rank-select rows of z into pooled (+bf16)
// ---------------------------------------------------------------------------
__global__ void k_topk(const float* __restrict__ scores, const float* __restrict__ z,
                       float* __restrict__ pooled, unsigned short* __restrict__ pooled_bf) {
  int b = blockIdx.x, i = threadIdx.x;
  if (i >= 225) return;
  const float* sb = scores + b * 225;
  float si = sb[i];
  int rank = 0;
  for (int j = 0; j < 225; ++j) {
    float sj = sb[j];
    rank += (sj > si) || (sj == si && j < i);
  }
  if (rank < 112) {
    const float* zr = z + ((size_t)b * 225 + i) * 128;
    size_t o = ((size_t)b * 112 + rank) * 128;
    for (int d = 0; d < 128; ++d) { pooled[o + d] = zr[d]; pooled_bf[o + d] = f2bf(zr[d]); }
  }
}

// ---------------------------------------------------------------------------
// 11) MHA attention scores + softmax (4 heads, hd=32, S=112)
// ---------------------------------------------------------------------------
__global__ void k_att(const float* __restrict__ qkv, float* __restrict__ att) {
  int idx = blockIdx.x * blockDim.x + threadIdx.x;    // 16*4*112
  if (idx >= 16 * 4 * 112) return;
  int q = idx % 112, bh = idx / 112;
  int h = bh & 3, b = bh >> 2;
  const float* qp = qkv + ((size_t)b * 112 + q) * 384 + h * 32;
  float sv[112];
  float mx = -__builtin_inff();
  for (int k = 0; k < 112; ++k) {
    const float* kp = qkv + ((size_t)b * 112 + k) * 384 + 128 + h * 32;
    float d = 0.0f;
    for (int e = 0; e < 32; ++e) d += qp[e] * kp[e];
    d *= 0.17677669529663687f;                        // 1/sqrt(32)
    sv[k] = d;
    mx = fmaxf(mx, d);
  }
  float sum = 0.0f;
  for (int k = 0; k < 112; ++k) { float e = expf(sv[k] - mx); sv[k] = e; sum += e; }
  float inv = 1.0f / sum;
  float* ap = att + (size_t)idx * 112;
  for (int k = 0; k < 112; ++k) ap[k] = sv[k] * inv;
}

// 12) o = att @ v (heads interleaved back to [b][s][128]), bf16 mirror out
__global__ void k_av(const float* __restrict__ att, const float* __restrict__ qkv,
                     float* __restrict__ o, unsigned short* __restrict__ o_bf) {
  int idx = blockIdx.x * blockDim.x + threadIdx.x;    // 16*112*128
  if (idx >= 16 * 112 * 128) return;
  int e = idx & 127, s = (idx >> 7) % 112, b = idx / (112 * 128);
  int h = e >> 5, d = e & 31;
  const float* ar = att + (((size_t)(b * 4 + h) * 112) + s) * 112;
  const float* vp = qkv + (size_t)b * 112 * 384 + 256 + h * 32 + d;
  float acc = 0.0f;
  for (int k = 0; k < 112; ++k) acc += ar[k] * vp[(size_t)k * 384];
  o[idx] = acc;
  o_bf[idx] = f2bf(acc);
}

// 13) layernorm over last dim (128), f32 out + bf16 mirror
__global__ void k_layernorm(const float* __restrict__ src, const float* __restrict__ g,
                            const float* __restrict__ bta, float* __restrict__ dst,
                            unsigned short* __restrict__ dst_bf, int rows) {
  int r = blockIdx.x * blockDim.x + threadIdx.x;
  if (r >= rows) return;
  const float* s = src + (size_t)r * 128;
  float m = 0.0f;
  for (int e = 0; e < 128; ++e) m += s[e];
  m *= (1.0f / 128.0f);
  float v = 0.0f;
  for (int e = 0; e < 128; ++e) { float t = s[e] - m; v += t * t; }
  v *= (1.0f / 128.0f);
  float inv = rsqrtf(v + 1e-5f);
  float* d = dst + (size_t)r * 128;
  for (int e = 0; e < 128; ++e) {
    float t = (s[e] - m) * inv * g[e] + bta[e];
    d[e] = t;
    if (dst_bf) dst_bf[(size_t)r * 128 + e] = f2bf(t);
  }
}

// 14) sequence-sum readout, f32 + bf16 mirror
__global__ void k_rep(const float* __restrict__ h, float* __restrict__ rep,
                      unsigned short* __restrict__ rep_bf) {
  int idx = blockIdx.x * blockDim.x + threadIdx.x;    // 16*128
  if (idx >= 16 * 128) return;
  int e = idx & 127, b = idx >> 7;
  const float* p = h + (size_t)b * 112 * 128 + e;
  float s = 0.0f;
  for (int k = 0; k < 112; ++k) s += p[(size_t)k * 128];
  rep[idx] = s;
  rep_bf[idx] = f2bf(s);
}

// ---------------------------------------------------------------------------
// launch
// ---------------------------------------------------------------------------
static inline int cdiv(int a, int b) { return (a + b - 1) / b; }

extern "C" void kernel_launch(void* const* d_in, const int* in_sizes, int n_in,
                              void* d_out, int out_size, void* d_ws, size_t ws_size,
                              hipStream_t stream) {
  const float* x        = (const float*)d_in[0];
  const float* conv1_w  = (const float*)d_in[1];
  const float* conv1_b  = (const float*)d_in[2];
  const float* bn1_g    = (const float*)d_in[3];
  const float* bn1_b    = (const float*)d_in[4];
  const float* conv2_w  = (const float*)d_in[5];
  const float* conv2_b  = (const float*)d_in[6];
  const float* bn2_g    = (const float*)d_in[7];
  const float* bn2_b    = (const float*)d_in[8];
  const float* proj_w   = (const float*)d_in[9];
  const float* proj_b   = (const float*)d_in[10];
  const float* w1_w     = (const float*)d_in[11];
  const float* w1_b     = (const float*)d_in[12];
  const float* w2_w     = (const float*)d_in[13];
  const float* w2_b     = (const float*)d_in[14];
  const float* gate_w   = (const float*)d_in[15];
  const float* gate_b   = (const float*)d_in[16];
  const float* score_w  = (const float*)d_in[17];
  const float* score_b  = (const float*)d_in[18];
  const float* in_pw    = (const float*)d_in[19];
  const float* in_pb    = (const float*)d_in[20];
  const float* out_pw   = (const float*)d_in[21];
  const float* out_pb   = (const float*)d_in[22];
  const float* ff1_w    = (const float*)d_in[23];
  const float* ff1_b    = (const float*)d_in[24];
  const float* ff2_w    = (const float*)d_in[25];
  const float* ff2_b    = (const float*)d_in[26];
  const float* ln1_g    = (const float*)d_in[27];
  const float* ln1_b    = (const float*)d_in[28];
  const float* ln2_g    = (const float*)d_in[29];
  const float* ln2_b    = (const float*)d_in[30];
  const float* cls1_w   = (const float*)d_in[31];
  const float* cls1_b   = (const float*)d_in[32];
  const float* cls2_w   = (const float*)d_in[33];
  const float* cls2_b   = (const float*)d_in[34];
  float* logits = (float*)d_out;

  // workspace carve-up
  char* p = (char*)d_ws;
  auto alloc = [&](size_t bytes) -> void* {
    void* r = (void*)p;
    p += (bytes + 255) & ~(size_t)255;
    return r;
  };
  unsigned short* wconv2   = (unsigned short*)alloc(36864 * 2);
  unsigned short* wproj    = (unsigned short*)alloc(524288 * 2);
  unsigned short* w1bf     = (unsigned short*)alloc(16384 * 2);
  unsigned short* w2bf     = (unsigned short*)alloc(16384 * 2);
  unsigned short* inpwbf   = (unsigned short*)alloc(49152 * 2);
  unsigned short* outpwbf  = (unsigned short*)alloc(16384 * 2);
  unsigned short* ff1wbf   = (unsigned short*)alloc(65536 * 2);
  unsigned short* ff2wbf   = (unsigned short*)alloc(65536 * 2);
  unsigned short* cls1wbf  = (unsigned short*)alloc(8192 * 2);
  unsigned short* feat1    = (unsigned short*)alloc((size_t)16 * 64 * 128 * 128 * 2);
  unsigned short* c2out    = (unsigned short*)alloc((size_t)16 * 64 * 128 * 128 * 2);
  unsigned short* feat2    = (unsigned short*)alloc((size_t)16 * 64 * 64 * 64 * 2);
  float* nodes   = (float*)alloc((size_t)16 * 225 * 128 * 4);
  float* adj1    = (float*)alloc(225 * 225 * 4);
  float* adj2    = (float*)alloc(225 * 225 * 4);
  unsigned short* agg1 = (unsigned short*)alloc((size_t)16 * 225 * 128 * 2);
  unsigned short* agg2 = (unsigned short*)alloc((size_t)16 * 225 * 128 * 2);
  float* z1      = (float*)alloc((size_t)16 * 225 * 128 * 4);
  float* z2      = (float*)alloc((size_t)16 * 225 * 128 * 4);
  float* zbuf    = (float*)alloc((size_t)16 * 225 * 128 * 4);
  float* scores  = (float*)alloc(16 * 225 * 4);
  float* pooled  = (float*)alloc((size_t)16 * 112 * 128 * 4);
  unsigned short* pooledbf = (unsigned short*)alloc((size_t)16 * 112 * 128 * 2);
  float* qkv     = (float*)alloc((size_t)16 * 112 * 384 * 4);
  float* att     = (float*)alloc((size_t)16 * 4 * 112 * 112 * 4);
  float* obuf    = (float*)alloc((size_t)16 * 112 * 128 * 4);
  unsigned short* obufbf = (unsigned short*)alloc((size_t)16 * 112 * 128 * 2);
  float* r1      = (float*)alloc((size_t)16 * 112 * 128 * 4);
  float* h1      = (float*)alloc((size_t)16 * 112 * 128 * 4);
  unsigned short* h1bf = (unsigned short*)alloc((size_t)16 * 112 * 128 * 2);
  float* ffbuf   = (float*)alloc((size_t)16 * 112 * 512 * 4);
  unsigned short* ffbufbf = (unsigned short*)alloc((size_t)16 * 112 * 512 * 2);
  float* r2      = (float*)alloc((size_t)16 * 112 * 128 * 4);
  float* h2      = (float*)alloc((size_t)16 * 112 * 128 * 4);
  float* rep     = (float*)alloc(16 * 128 * 4);
  unsigned short* repbf = (unsigned short*)alloc(16 * 128 * 2);
  float* c1buf   = (float*)alloc(16 * 64 * 4);
  (void)ws_size; (void)n_in; (void)in_sizes; (void)out_size;

  // 0) weights -> bf16
  k_tobf16<<<cdiv(36864, 256), 256, 0, stream>>>(conv2_w, wconv2, 36864);
  k_tobf16<<<cdiv(524288, 256), 256, 0, stream>>>(proj_w, wproj, 524288);
  k_tobf16<<<cdiv(16384, 256), 256, 0, stream>>>(w1_w, w1bf, 16384);
  k_tobf16<<<cdiv(16384, 256), 256, 0, stream>>>(w2_w, w2bf, 16384);
  k_tobf16<<<cdiv(49152, 256), 256, 0, stream>>>(in_pw, inpwbf, 49152);
  k_tobf16<<<cdiv(16384, 256), 256, 0, stream>>>(out_pw, outpwbf, 16384);
  k_tobf16<<<cdiv(65536, 256), 256, 0, stream>>>(ff1_w, ff1wbf, 65536);
  k_tobf16<<<cdiv(65536, 256), 256, 0, stream>>>(ff2_w, ff2wbf, 65536);
  k_tobf16<<<cdiv(8192, 256), 256, 0, stream>>>(cls1_w, cls1wbf, 8192);

  // 1) conv1+bn+relu+pool
  k_conv1_bn_relu_pool<<<cdiv(16 * 64 * 128 * 128, 256), 256, 0, stream>>>(
      x, conv1_w, conv1_b, bn1_g, bn1_b, feat1);
  // 2) conv2 implicit-GEMM WMMA (16384 waves, LDS-staged weights)
  k_conv2_wmma<<<4096, 128, 0, stream>>>(feat1, wconv2, conv2_b, bn2_g, bn2_b, c2out);
  // 3) pool -> feat2
  k_pool2<<<cdiv(16 * 64 * 64 * 64, 256), 256, 0, stream>>>(c2out, feat2);
  // 4) patch projection WMMA (225 waves, LDS-staged weights)
  k_proj_wmma<<<cdiv(225, 4), 128, 0, stream>>>(feat2, wproj, proj_b, nodes);
  // 5) adjacency
  k_adj1<<<1, 256, 0, stream>>>(adj1);
  k_adj2<<<225, 256, 0, stream>>>(adj1, adj2);
  // 6) aggregation
  k_aggregate<<<cdiv(16 * 225 * 128, 256), 256, 0, stream>>>(adj1, nodes, agg1);
  k_aggregate<<<cdiv(16 * 225 * 128, 256), 256, 0, stream>>>(adj2, nodes, agg2);
  // 7) GCN transforms via generic WMMA linear (3600x128 @ 128x128)
  k_linear_wmma<<<cdiv(225 * 8, 4), 128, 0, stream>>>(agg1, w1bf, w1_b, nullptr,
                                                      z1, nullptr, 3600, 128, 128, 0);
  k_linear_wmma<<<cdiv(225 * 8, 4), 128, 0, stream>>>(agg2, w2bf, w2_b, nullptr,
                                                      z2, nullptr, 3600, 128, 128, 0);
  // 8) gating
  k_gate<<<cdiv(16 * 225 * 128, 256), 256, 0, stream>>>(z1, z2, gate_w, gate_b, zbuf);
  // 9) scores (out_dim=1, VALU)
  k_linear<<<cdiv(3600, 256), 256, 0, stream>>>(zbuf, score_w, score_b, scores,
                                                3600, 128, 1);
  // 10) top-k pooling (writes f32 + bf16)
  k_topk<<<16, 256, 0, stream>>>(scores, zbuf, pooled, pooledbf);
  // 11) MHA: qkv via WMMA (1792x128 @ 128x384)
  k_linear_wmma<<<cdiv(112 * 24, 4), 128, 0, stream>>>(pooledbf, inpwbf, in_pb, nullptr,
                                                       qkv, nullptr, 1792, 128, 384, 0);
  k_att<<<cdiv(16 * 4 * 112, 128), 128, 0, stream>>>(qkv, att);
  k_av<<<cdiv(1792 * 128, 256), 256, 0, stream>>>(att, qkv, obuf, obufbf);
  // out-proj via WMMA with residual(pooled)
  k_linear_wmma<<<cdiv(112 * 8, 4), 128, 0, stream>>>(obufbf, outpwbf, out_pb, pooled,
                                                      r1, nullptr, 1792, 128, 128, 0);
  k_layernorm<<<cdiv(1792, 256), 256, 0, stream>>>(r1, ln1_g, ln1_b, h1, h1bf, 1792);
  // 12) FFN via WMMA
  k_linear_wmma<<<cdiv(112 * 32, 4), 128, 0, stream>>>(h1bf, ff1wbf, ff1_b, nullptr,
                                                       ffbuf, ffbufbf, 1792, 128, 512, 1);
  k_linear_wmma<<<cdiv(112 * 8, 4), 128, 0, stream>>>(ffbufbf, ff2wbf, ff2_b, h1,
                                                      r2, nullptr, 1792, 512, 128, 0);
  k_layernorm<<<cdiv(1792, 256), 256, 0, stream>>>(r2, ln2_g, ln2_b, h2, nullptr, 1792);
  // 13) readout + classifier (cls1 via WMMA: 16x128 @ 128x64)
  k_rep<<<cdiv(16 * 128, 256), 256, 0, stream>>>(h2, rep, repbf);
  k_linear_wmma<<<1, 128, 0, stream>>>(repbf, cls1wbf, cls1_b, nullptr,
                                       c1buf, nullptr, 16, 128, 64, 1);
  k_linear<<<1, 256, 0, stream>>>(c1buf, cls2_w, cls2_b, logits, 16, 64, 2);
}
